// GAT_33251636806286
// MI455X (gfx1250) — matrix operand order repaired and checked by reading, not compiled, not run on hardware
//
#include <hip/hip_runtime.h>
#include <hip/hip_fp16.h>

// GAT forward for MI455X (gfx1250).
// - All matmuls on v_wmma_f32_16x16x32_f16 (f16 operands, f32 accumulate).
// - Flash-style fused masked softmax + P@V, never materializing [H,N,N].
// - Stationary operands kept transposed in global so B-fragments are 2x b128 loads.
// - Shared tiles (adj, A, f2) double-buffered via global_load_async_to_lds_b128,
//   drained with s_wait_asynccnt N (async completes in order; every wave issues
//   the same number of async ops per tile so the wait count is wave-uniform).

#define NND 4096   // nodes
#define NFT 1024   // input features
#define NHD 128    // hidden per head
#define NH  8      // heads
#define NC  64     // classes
#define GAT_ALPHA 0.2f
#define NEGINF (-9.0e15f)

typedef __attribute__((ext_vector_type(16))) _Float16 v16h;
typedef __attribute__((ext_vector_type(8)))  float    v8f;

union HFrag { v16h v; uint4 q[2]; };       // 16 halfs = 2 x 16B
union I16  { int4   q4[4]; int   a[16]; };
union F16v { float4 q4[4]; float a[16]; };

__device__ __forceinline__ float lrelu(float x) { return x > 0.f ? x : GAT_ALPHA * x; }
__device__ __forceinline__ float eluf(float x)  { return x > 0.f ? x : (__expf(x) - 1.f); }

// Async memory->LDS copy, 16B per active lane (tracked by ASYNCcnt).
__device__ __forceinline__ void async_b128(void* lds_dst, const void* gsrc) {
    unsigned lds = (unsigned)(size_t)lds_dst;  // low 32 bits of generic LDS ptr = LDS offset
    asm volatile("global_load_async_to_lds_b128 %0, %1, off"
                 :: "v"(lds), "v"(gsrc) : "memory");
}
template <int N>
__device__ __forceinline__ void wait_async() {
    if constexpr (N == 0)      asm volatile("s_wait_asynccnt 0x0" ::: "memory");
    else if constexpr (N == 1) asm volatile("s_wait_asynccnt 0x1" ::: "memory");
    else                       asm volatile("s_wait_asynccnt 0x2" ::: "memory");
}

// ---------------------------------------------------------------- converts
__global__ void cvt_f32_f16(const float* __restrict__ s, _Float16* __restrict__ d, int n) {
    int i = blockIdx.x * blockDim.x + threadIdx.x;
    if (i < n) d[i] = (_Float16)s[i];
}

// src[b][R][C] f32 -> dst[b][C][R] f16 (coalesced read; done once, tiny)
__global__ void cvtT_f32_f16(const float* __restrict__ s, _Float16* __restrict__ d,
                             int B, int R, int C) {
    long i   = (long)blockIdx.x * blockDim.x + threadIdx.x;
    long tot = (long)B * R * C;
    if (i >= tot) return;
    int b   = (int)(i / ((long)R * C));
    int rem = (int)(i % ((long)R * C));
    int r = rem / C, c = rem % C;
    d[(size_t)b * R * C + (size_t)c * R + r] = (_Float16)s[i];
}

// ---------------------------------------------------------------- WMMA GEMM
// CT[head] = (A[head] @ Bt[head]^T)^T.  A:[M,KDIM] row-major f16, Bt:[NDIM,KDIM]
// f16 (pre-transposed), CT:[NDIM,M] f16 (transposed out, feeds attention V^T).
// One 16-row strip per workgroup; wave w owns n-tile w. A tile double-buffered in
// LDS via async copy (1 async/tile/wave); B straight from global, 2 x b128/lane.
template <int KDIM, int NDIM, int NWAVES>
__global__ __launch_bounds__(NWAVES * 32)
void gemm_wmma(const _Float16* __restrict__ A, size_t strideAh,
               const _Float16* __restrict__ Bt, size_t strideBh,
               _Float16* __restrict__ CT, size_t strideCTh) {
    constexpr int ASEG = 64 / NWAVES;  // 16B segments of the A tile per wave
    __shared__ __align__(16) _Float16 sA[2][16 * 32];
    const int head = blockIdx.y;
    const int m0   = blockIdx.x * 16;
    const int wave = threadIdx.x >> 5;
    const int lane = threadIdx.x & 31;
    const int n0   = wave * 16;
    const int mrow = lane & 15, kh = lane >> 4, n = lane & 15;
    const _Float16* Ah  = A + (size_t)head * strideAh;
    const _Float16* Btn = Bt + (size_t)head * strideBh + (size_t)(n0 + n) * KDIM;

    auto issueA = [&](int k0, int buf) {  // every wave issues exactly 1 async op
        if (lane < ASEG) {
            int seg = wave * ASEG + lane;
            int r = seg >> 2, cq = seg & 3;
            async_b128(&sA[buf][r * 32 + cq * 8],
                       &Ah[(size_t)(m0 + r) * KDIM + k0 + cq * 8]);
        }
    };

    v8f acc = {};
    issueA(0, 0);
    for (int k0 = 0; k0 < KDIM; k0 += 32) {
        const int buf = (k0 >> 5) & 1;
        if (k0 + 32 < KDIM) { issueA(k0 + 32, buf ^ 1); wait_async<1>(); }
        else                { wait_async<0>(); }
        __syncthreads();
        HFrag a, b;
        a.q[0] = *(const uint4*)&sA[buf][mrow * 32 + kh * 8];       // K = kh*8 .. +7
        a.q[1] = *(const uint4*)&sA[buf][mrow * 32 + 16 + kh * 8];  // K = 16+kh*8 .. +7
        b.q[0] = *(const uint4*)&Btn[k0 + 8 * kh];                  // K = kh*8 .. +7
        b.q[1] = *(const uint4*)&Btn[k0 + 16 + 8 * kh];             // K = 16+kh*8 .. +7
        acc = __builtin_amdgcn_wmma_f32_16x16x32_f16(false, a.v, false, b.v, (short)0,
                                                     acc, false, false);
        __syncthreads();  // all waves done with buf before it is refilled
    }
#pragma unroll
    for (int v = 0; v < 8; ++v) {
        int row = v + 8 * kh;
        CT[(size_t)head * strideCTh + (size_t)(n0 + n) * NND + (m0 + row)] = (_Float16)acc[v];
    }
}

// ---------------------------------------------------------------- score vectors
// WhT:[H][D][N] -> f1/f2:[H][N].  Adjacent threads read adjacent nodes: coalesced.
__global__ void fscore_kernel(const _Float16* __restrict__ WhT, const float* __restrict__ a1,
                              const float* __restrict__ a2, float* f1, float* f2) {
    int idx = blockIdx.x * blockDim.x + threadIdx.x;
    if (idx >= NH * NND) return;
    int head = idx / NND, j = idx % NND;
    const _Float16* col = WhT + (size_t)head * NND * NHD + j;
    const float* A1 = a1 + head * NHD;
    const float* A2 = a2 + head * NHD;
    float s1 = 0.f, s2 = 0.f;
    for (int d = 0; d < NHD; ++d) {
        float v = (float)col[(size_t)d * NND];
        s1 += v * A1[d];
        s2 += v * A2[d];
    }
    f1[idx] = s1;
    f2[idx] = s2;
}

__global__ void gscore_kernel(const _Float16* __restrict__ WhoT, const float* __restrict__ ao1,
                              const float* __restrict__ ao2, float* g1, float* g2) {
    int j = blockIdx.x * blockDim.x + threadIdx.x;
    if (j >= NND) return;
    float s1 = 0.f, s2 = 0.f;
    for (int c = 0; c < NC; ++c) {
        float v = (float)WhoT[(size_t)c * NND + j];
        s1 += v * ao1[c];
        s2 += v * ao2[c];
    }
    g1[j] = s1;
    g2[j] = s2;
}

// ---------------------------------------------------------------- fused attention
// One workgroup = (head, 16-row block); NWAVES waves each own a 16-wide d-tile of
// D = 16*NWAVES. Streams 32-column chunks: masked LeakyReLU scores -> online
// softmax -> WMMA P@V. V^T:[H][D][N] so B-fragments are 2 x b128 global loads.
// adj + f2 tiles double-buffered via async copy (2 asyncs/tile/wave).
// out16 != null: ELU + f16 concat write (layer 1). out32: ELU + f32 (layer 2).
template <int NWAVES>
__global__ __launch_bounds__(NWAVES * 32)
void attn_wmma(const int* __restrict__ adj, const float* __restrict__ f_self,
               const float* __restrict__ f_nbr, const _Float16* __restrict__ VT,
               _Float16* __restrict__ out16, float* __restrict__ out32, int outLD) {
    constexpr int D    = 16 * NWAVES;
    constexpr int ASEG = 128 / NWAVES;  // adj 16B segments per wave
    constexpr int FSEG = 8 / NWAVES;    // f2  16B segments per wave
    __shared__ __align__(16) int   sAdj[2][16 * 32];
    __shared__ __align__(16) float sFn[2][32];
    const int head = blockIdx.y;
    const int m0   = blockIdx.x * 16;
    const int wave = threadIdx.x >> 5;
    const int lane = threadIdx.x & 31;
    const int n0   = wave * 16;
    const int mrow = lane & 15, kh = lane >> 4, n = lane & 15;
    const float fs       = f_self[head * NND + m0 + mrow];
    const float* Fn      = f_nbr + head * NND;
    const _Float16* vrow = VT + ((size_t)head * D + n0 + n) * NND;

    auto issue = [&](int j0, int buf) {  // every wave issues exactly 2 async ops
        if (lane < ASEG) {
            int seg = wave * ASEG + lane;
            int r = seg >> 3, cq = seg & 7;
            async_b128(&sAdj[buf][r * 32 + cq * 4],
                       &adj[(size_t)(m0 + r) * NND + j0 + cq * 4]);
        }
        if (lane >= 16 && lane < 16 + FSEG) {
            int seg = wave * FSEG + (lane - 16);
            async_b128(&sFn[buf][seg * 4], &Fn[j0 + seg * 4]);
        }
    };

    float rmax = -INFINITY, rsum = 0.f;
    v8f acc = {};
    issue(0, 0);
    for (int j0 = 0; j0 < NND; j0 += 32) {
        const int buf = (j0 >> 5) & 1;
        if (j0 + 32 < NND) { issue(j0 + 32, buf ^ 1); wait_async<2>(); }
        else               { wait_async<0>(); }
        __syncthreads();
        // vectorized score reads: ka = (i&7) + ((i>>3)<<4) + kh*8
        I16  ad;
        F16v fn;
        const int ab = mrow * 32 + kh * 8;
        ad.q4[0] = *(const int4*)&sAdj[buf][ab];
        ad.q4[1] = *(const int4*)&sAdj[buf][ab + 4];
        ad.q4[2] = *(const int4*)&sAdj[buf][ab + 16];
        ad.q4[3] = *(const int4*)&sAdj[buf][ab + 20];
        const int fb = kh * 8;
        fn.q4[0] = *(const float4*)&sFn[buf][fb];
        fn.q4[1] = *(const float4*)&sFn[buf][fb + 4];
        fn.q4[2] = *(const float4*)&sFn[buf][fb + 16];
        fn.q4[3] = *(const float4*)&sFn[buf][fb + 20];
        float e[16];
        float cmax = -INFINITY;
#pragma unroll
        for (int i = 0; i < 16; ++i) {
            float sc = ad.a[i] > 0 ? lrelu(fs + fn.a[i]) : NEGINF;
            e[i]     = sc;
            cmax     = fmaxf(cmax, sc);
        }
        cmax        = fmaxf(cmax, __shfl_xor(cmax, 16, 32));
        float nmax  = fmaxf(rmax, cmax);
        float scale = __expf(rmax - nmax);
        HFrag a;
        float psum = 0.f;
#pragma unroll
        for (int i = 0; i < 16; ++i) {
            float p = __expf(e[i] - nmax);
            a.v[i]  = (_Float16)p;
            psum += p;
        }
        psum += __shfl_xor(psum, 16, 32);
        rsum = rsum * scale + psum;
        rmax = nmax;
#pragma unroll
        for (int v = 0; v < 8; ++v) acc[v] *= __shfl(scale, v + 8 * kh, 32);
        HFrag b;
        b.q[0] = *(const uint4*)&vrow[j0 + 8 * kh];
        b.q[1] = *(const uint4*)&vrow[j0 + 16 + 8 * kh];
        acc = __builtin_amdgcn_wmma_f32_16x16x32_f16(false, a.v, false, b.v, (short)0,
                                                     acc, false, false);
        __syncthreads();  // all waves done with buf before it is refilled
    }
#pragma unroll
    for (int v = 0; v < 8; ++v) {
        int row    = v + 8 * kh;
        float rs   = __shfl(rsum, row, 32);
        float val  = eluf(acc[v] / rs);
        size_t idx = (size_t)(m0 + row) * outLD + head * D + n0 + n;
        if (out16) out16[idx] = (_Float16)val;
        else       out32[idx] = val;
    }
}

// ---------------------------------------------------------------- log_softmax
__global__ void logsoftmax_kernel(const float* __restrict__ O, float* __restrict__ out) {
    int r = blockIdx.x * blockDim.x + threadIdx.x;
    if (r >= NND) return;
    const float* row = O + (size_t)r * NC;
    float m = -INFINITY;
    for (int c = 0; c < NC; ++c) m = fmaxf(m, row[c]);
    float s = 0.f;
    for (int c = 0; c < NC; ++c) s += __expf(row[c] - m);
    float lg = m + __logf(s);
    for (int c = 0; c < NC; ++c) out[(size_t)r * NC + c] = row[c] - lg;
}

// ---------------------------------------------------------------- host launcher
extern "C" void kernel_launch(void* const* d_in, const int* in_sizes, int n_in, void* d_out,
                              int out_size, void* d_ws, size_t ws_size, hipStream_t stream) {
    (void)in_sizes; (void)n_in; (void)out_size; (void)ws_size;
    const float* x   = (const float*)d_in[0];
    const int*   adj = (const int*)d_in[1];
    const float* W   = (const float*)d_in[2];
    const float* a1  = (const float*)d_in[3];
    const float* a2  = (const float*)d_in[4];
    const float* Wo  = (const float*)d_in[5];
    const float* ao1 = (const float*)d_in[6];
    const float* ao2 = (const float*)d_in[7];
    float* out = (float*)d_out;

    size_t off = 0;
    auto carve = [&](size_t bytes) -> void* {
        void* p = (char*)d_ws + off;
        off += (bytes + 255) & ~(size_t)255;
        return p;
    };
    _Float16* xh     = (_Float16*)carve((size_t)NND * NFT * 2);       // 8 MB
    _Float16* wWT    = (_Float16*)carve((size_t)NH * NHD * NFT * 2);  // 2 MB  W^T[h][d][k]
    _Float16* wWoT   = (_Float16*)carve((size_t)NC * NFT * 2);        //       Wo^T[c][k]
    _Float16* WhT16  = (_Float16*)carve((size_t)NH * NHD * NND * 2);  // 8 MB  Wh^T[h][d][j]
    float*    f1     = (float*)carve((size_t)NH * NND * 4);
    float*    f2     = (float*)carve((size_t)NH * NND * 4);
    _Float16* hcat   = (_Float16*)carve((size_t)NND * NH * NHD * 2);  // 8 MB  [node][h*D+d]
    _Float16* WhoT16 = (_Float16*)carve((size_t)NC * NND * 2);        //       Who^T[c][j]
    float*    g1     = (float*)carve((size_t)NND * 4);
    float*    g2     = (float*)carve((size_t)NND * 4);
    float*    O32    = (float*)carve((size_t)NND * NC * 4);

    cvt_f32_f16<<<(NND * NFT + 255) / 256, 256, 0, stream>>>(x, xh, NND * NFT);
    cvtT_f32_f16<<<(NH * NFT * NHD + 255) / 256, 256, 0, stream>>>(W, wWT, NH, NFT, NHD);
    cvtT_f32_f16<<<(NFT * NC + 255) / 256, 256, 0, stream>>>(Wo, wWoT, 1, NFT, NC);

    // Wh^T[h] = (x @ W[h])^T
    gemm_wmma<NFT, NHD, 8><<<dim3(NND / 16, NH), 256, 0, stream>>>(
        xh, 0, wWT, (size_t)NHD * NFT, WhT16, (size_t)NHD * NND);
    fscore_kernel<<<(NH * NND + 255) / 256, 256, 0, stream>>>(WhT16, a1, a2, f1, f2);
    // layer 1: fused masked softmax + attn@Wh + ELU + head concat
    attn_wmma<8><<<dim3(NND / 16, NH), 256, 0, stream>>>(adj, f1, f2, WhT16, hcat, nullptr,
                                                         NH * NHD);
    // Who^T = (hcat @ Wo)^T
    gemm_wmma<NFT, NC, 4><<<dim3(NND / 16, 1), 128, 0, stream>>>(
        hcat, 0, wWoT, 0, WhoT16, 0);
    gscore_kernel<<<(NND + 255) / 256, 256, 0, stream>>>(WhoT16, ao1, ao2, g1, g2);
    // layer 2: fused masked softmax + ao@Who + ELU
    attn_wmma<4><<<dim3(NND / 16, 1), 128, 0, stream>>>(adj, g1, g2, WhoT16, nullptr, O32, NC);
    logsoftmax_kernel<<<(NND + 255) / 256, 256, 0, stream>>>(O32, out);
}
